// ProposedModel_41583873360047
// MI455X (gfx1250) — matrix operand order repaired and checked by reading, not compiled, not run on hardware
//
#include <hip/hip_runtime.h>
#include <cstdint>
#include <cstddef>

// ---------------------------------------------------------------------------
// Types for CDNA5 WMMA (wave32, V_WMMA_F32_16X16X32_BF16)
// ---------------------------------------------------------------------------
typedef __bf16 bf16;
typedef __attribute__((ext_vector_type(16))) __bf16 v16bf;
typedef __attribute__((ext_vector_type(8)))  __bf16 v8bf;
typedef __attribute__((ext_vector_type(8)))  float   v8f;

#define LNUM 6
#define BB   4
#define TT   512
#define DM   768
#define HH   8
#define DQ   32
#define FF   3072
#define OUTN 80
#define BT   (BB*TT)     // 2048 rows of activations
#define NREL 199
#define NRELP 208        // 199 padded to multiple of 16

__device__ __forceinline__ v8f wmma_bf16(v16bf a, v16bf b, v8f c) {
  // (neg_a, A, neg_b, B, c_mod, C, reuse_a, reuse_b)
  return __builtin_amdgcn_wmma_f32_16x16x32_bf16(false, a, false, b, (short)0, c,
                                                 false, false);
}

// CDNA5 async memory->LDS DMA (tracked by ASYNCcnt). LDS destination address
// is the low 32 bits of the generic pointer (flat->LDS aperture: addr[31:0]).
__device__ __forceinline__ void async_b128(const bf16* g, bf16* l) {
  asm volatile("global_load_async_to_lds_b128 %0, %1, off"
               :: "v"((uint32_t)(uintptr_t)l), "v"((uint64_t)(uintptr_t)g)
               : "memory");
}
__device__ __forceinline__ void wait_async() {
  asm volatile("s_wait_asynccnt 0x0" ::: "memory");
}

// Build a 16-element bf16 fragment from two contiguous 8-element chunks.
// A-frag: lane<16 holds K[0..15]; lane>=16 holds K[16..31]  (M = lane&15)
// B-frag: lane<16 holds K[0..15]; lane>=16 holds K[16..31]  (N = lane&15)
__device__ __forceinline__ v16bf frag_cat(const bf16* lo, const bf16* hi) {
  v8bf l = *(const v8bf*)lo;
  v8bf h = *(const v8bf*)hi;
  v16bf r;
#pragma unroll
  for (int i = 0; i < 8; ++i) { r[i] = l[i]; r[i + 8] = h[i]; }
  return r;
}

// ---------------------------------------------------------------------------
// Weight packing kernels: f32 -> bf16 in [N][K] layout for B-fragment loads
// ---------------------------------------------------------------------------
// Fused QKV weight: [l][n(768)][f] where rows 0-255=Q, 256-511=K, 512-767=V,
// within each block n%256 = h*32+a.
__global__ void pack_qkv_kernel(const float* __restrict__ wq,
                                const float* __restrict__ wk,
                                const float* __restrict__ wv,
                                bf16* __restrict__ pqkv) {
  const size_t n = (size_t)LNUM * 768 * DM;
  for (size_t idx = (size_t)blockIdx.x * 256 + threadIdx.x; idx < n;
       idx += (size_t)gridDim.x * 256) {
    int f = (int)(idx % DM);
    size_t t = idx / DM;
    int nr = (int)(t % 768);
    int l = (int)(t / 768);
    int sel = nr >> 8;            // 0=Q, 1=K, 2=V
    int c = nr & 255;
    int hh = c >> 5, a = c & 31;
    size_t src = (((size_t)l * HH + hh) * DM + f) * DQ + a;  // (L,H,DM,DQ)
    const float* w = (sel == 0) ? wq : ((sel == 1) ? wk : wv);
    pqkv[idx] = (bf16)w[src];
  }
}

__global__ void pack_wo_kernel(const float* __restrict__ wo,
                               bf16* __restrict__ po) {
  const size_t n = (size_t)LNUM * DM * 256;   // [l][d][c=h*32+a]
  for (size_t idx = (size_t)blockIdx.x * 256 + threadIdx.x; idx < n;
       idx += (size_t)gridDim.x * 256) {
    int c = (int)(idx % 256);
    size_t t = idx / 256;
    int d = (int)(t % DM);
    int l = (int)(t / DM);
    int hh = c >> 5, a = c & 31;
    size_t src = (((size_t)l * HH + hh) * DQ + a) * DM + d;  // (L,H,DQ,DM)
    po[idx] = (bf16)wo[src];
  }
}

__global__ void pack_rel_kernel(const float* __restrict__ rel,
                                bf16* __restrict__ pr) {
  const size_t n = (size_t)LNUM * HH * NRELP * DQ;  // [l][h][m(208)][a]
  for (size_t idx = (size_t)blockIdx.x * 256 + threadIdx.x; idx < n;
       idx += (size_t)gridDim.x * 256) {
    int a = (int)(idx % DQ);
    size_t t = idx / DQ;
    int m = (int)(t % NRELP);
    size_t t2 = t / NRELP;
    int hh = (int)(t2 % HH);
    int l = (int)(t2 / HH);
    pr[idx] = (m < NREL)
                  ? (bf16)rel[(((size_t)l * HH + hh) * NREL + m) * DQ + a]
                  : (bf16)0.0f;
  }
}

__global__ void conv_kernel(const float* __restrict__ src,
                            bf16* __restrict__ dst, size_t n) {
  for (size_t i = (size_t)blockIdx.x * 256 + threadIdx.x; i < n;
       i += (size_t)gridDim.x * 256)
    dst[i] = (bf16)src[i];
}

__global__ void pack_wout_kernel(const float* __restrict__ w,
                                 bf16* __restrict__ p) {
  const size_t n = (size_t)128 * DM;   // [n(128, 80 valid)][f]
  for (size_t idx = (size_t)blockIdx.x * 256 + threadIdx.x; idx < n;
       idx += (size_t)gridDim.x * 256) {
    int f = (int)(idx % DM);
    int o = (int)(idx / DM);
    p[idx] = (o < OUTN) ? (bf16)w[(size_t)o * DM + f] : (bf16)0.0f;
  }
}

__global__ void init_h_kernel(const float* __restrict__ x,
                              float* __restrict__ h, bf16* __restrict__ xb,
                              size_t n) {
  for (size_t i = (size_t)blockIdx.x * 256 + threadIdx.x; i < n;
       i += (size_t)gridDim.x * 256) {
    float v = x[i];
    h[i] = v;
    xb[i] = (bf16)v;
  }
}

// ---------------------------------------------------------------------------
// Generic bf16 WMMA GEMM:  C[M][N] = A[M][K] x Bt[N][K]^T (+bias, relu)
// Workgroup tile 128x128, 8 waves in 4x2, wave tile 32x64 (2x4 WMMA frags).
// Double-buffered LDS tiles filled by GLOBAL_LOAD_ASYNC_TO_LDS_B128, so the
// DMA for tile k+1 overlaps the 64 WMMAs of tile k; sync via s_wait_asynccnt.
// ---------------------------------------------------------------------------
__global__ __launch_bounds__(256)
void gemm_bf16_kernel(const bf16* __restrict__ A, int lda,
                      const bf16* __restrict__ Bt, int ldb,
                      float* __restrict__ Cf, bf16* __restrict__ Cb, int ldc,
                      int Nstore, int K, const float* __restrict__ bias,
                      int relu) {
  __shared__ bf16 sA[2][128 * 40];   // 128 rows x 32 K, padded stride 40
  __shared__ bf16 sB[2][128 * 40];   // 128 N-rows x 32 K, padded stride 40
  const int tid = threadIdx.x, lane = tid & 31, wid = tid >> 5;
  const int l15 = lane & 15, hi = lane >> 4;
  const int mBlk = blockIdx.y * 128, nBlk = blockIdx.x * 128;
  const int mW = (wid & 3) * 32;     // 4 row-waves
  const int nW = (wid >> 2) * 64;    // 2 col-waves
  const int ldr = tid >> 1, ldc16 = (tid & 1) * 16;  // 128 rows x 2 halves
  const bf16* Ap = A + (size_t)(mBlk + ldr) * lda + ldc16;
  const bf16* Bp = Bt + (size_t)(nBlk + ldr) * ldb + ldc16;
  bf16* dA = &sA[0][ldr * 40 + ldc16];
  bf16* dB = &sB[0][ldr * 40 + ldc16];

  auto issue = [&](int bufSel, int kk) {
    bf16* a = dA + bufSel * 128 * 40;
    bf16* b = dB + bufSel * 128 * 40;
    async_b128(Ap + kk, a);
    async_b128(Ap + kk + 8, a + 8);
    async_b128(Bp + kk, b);
    async_b128(Bp + kk + 8, b + 8);
  };

  v8f acc[2][4] = {};
  issue(0, 0);
  int buf = 0;
  for (int k0 = 0; k0 < K; k0 += 32) {
    wait_async();        // this wave's DMA into sX[buf] complete
    __syncthreads();     // all waves' DMA complete; prev tile fully consumed
    if (k0 + 32 < K) issue(buf ^ 1, k0 + 32);

    const bf16* sAb = sA[buf];
    const bf16* sBb = sB[buf];
    v16bf af[2], bfr[4];
#pragma unroll
    for (int i = 0; i < 2; ++i) {
      const bf16* row = sAb + (mW + i * 16 + l15) * 40;
      af[i] = frag_cat(row + hi * 16, row + hi * 16 + 8);
    }
#pragma unroll
    for (int j = 0; j < 4; ++j) {
      const bf16* bp = sBb + (nW + j * 16 + l15) * 40 + hi * 16;
      bfr[j] = frag_cat(bp, bp + 8);
    }
#pragma unroll
    for (int i = 0; i < 2; ++i)
#pragma unroll
      for (int j = 0; j < 4; ++j)
        acc[i][j] = wmma_bf16(af[i], bfr[j], acc[i][j]);
    buf ^= 1;
  }

#pragma unroll
  for (int i = 0; i < 2; ++i)
#pragma unroll
    for (int j = 0; j < 4; ++j) {
      const int col = nBlk + nW + j * 16 + l15;
      const float bs = (bias && col < Nstore) ? bias[col] : 0.0f;
#pragma unroll
      for (int r = 0; r < 8; ++r) {
        const int row = mBlk + mW + i * 16 + r + hi * 8;
        float v = acc[i][j][r] + bs;
        if (relu) v = fmaxf(v, 0.0f);
        if (col < Nstore) {
          if (Cf) Cf[(size_t)row * ldc + col] = v;
          else    Cb[(size_t)row * ldc + col] = (bf16)v;
        }
      }
    }
}

// ---------------------------------------------------------------------------
// Attention kernel: one workgroup per (b, h, 32 query rows).
// qkv input is the fused [BT][768] bf16 buffer (Q cols 0-255, K 256-511,
// V 512-767; within a block, head h occupies cols h*32..h*32+31).
// LDS layout (bytes):
//   [0,40960)       K tile [512][40] bf16, later overlaid with probs [32][520]
//   [40960,74240)   V^T [32][520] bf16
//   [74240,76800)   Q [32][40] bf16 (scaled by 1/sqrt(DQ) at staging)
//   [76800,103424)  rel_dot [32][208] f32
//   [103424,105472) reduction scratch max/sum [2][8][32] f32
//   [105472,109568) O accumulator [32][32] f32
// ---------------------------------------------------------------------------
#define SMEM_ATTN 109568
#define QKVLD 768

__global__ __launch_bounds__(256)
void attn_kernel(const bf16* __restrict__ qkv, const bf16* __restrict__ prel_l,
                 bf16* __restrict__ ob, float qscale) {
  extern __shared__ char smem[];
  bf16*  sKP   = (bf16*)(smem);                 // [512][40]; probs [32][520]
  bf16*  sVT   = (bf16*)(smem + 40960);         // [32][520]
  bf16*  sQ    = (bf16*)(smem + 74240);         // [32][40]
  float* sRel  = (float*)(smem + 76800);        // [32][208]
  float* sRedM = (float*)(smem + 103424);       // [8][32]
  float* sRedS = sRedM + 256;                   // [8][32]
  float* sO    = (float*)(smem + 105472);       // [32][32]

  const int tid = threadIdx.x, lane = tid & 31, wid = tid >> 5;
  const int l15 = lane & 15, hi = lane >> 4;
  const int bh = blockIdx.y;
  const int b = bh / HH, h = bh % HH;
  const int qBase = blockIdx.x * 32;

  const bf16* qHead = qkv + (size_t)(b * TT + qBase) * QKVLD + h * DQ;
  const bf16* kHead = qkv + (size_t)b * TT * QKVLD + 256 + h * DQ;
  const bf16* vHead = qkv + (size_t)b * TT * QKVLD + 512 + h * DQ;

  // Stage K rows via async DMA; V transposed manually (needs scatter).
  for (int r = tid; r < TT; r += 256) {
    const bf16* ks = kHead + (size_t)r * QKVLD;
    bf16* kd = sKP + r * 40;
    async_b128(ks, kd);
    async_b128(ks + 8, kd + 8);
    async_b128(ks + 16, kd + 16);
    async_b128(ks + 24, kd + 24);
    const v8bf* vs = (const v8bf*)(vHead + (size_t)r * QKVLD);
    v8bf v0 = vs[0], v1 = vs[1], v2 = vs[2], v3 = vs[3];
#pragma unroll
    for (int a = 0; a < 8; ++a) {
      sVT[(a     ) * 520 + r] = v0[a];
      sVT[(a +  8) * 520 + r] = v1[a];
      sVT[(a + 16) * 520 + r] = v2[a];
      sVT[(a + 24) * 520 + r] = v3[a];
    }
  }
  if (tid < 128) {   // Q staging with folded 1/sqrt(DQ)
    int r = tid >> 2, c = (tid & 3) * 8;
    v8bf qv = *(const v8bf*)(qHead + (size_t)r * QKVLD + c);
#pragma unroll
    for (int i = 0; i < 8; ++i) qv[i] = (bf16)((float)qv[i] * qscale);
    *(v8bf*)(sQ + r * 40 + c) = qv;
  }
  wait_async();
  __syncthreads();

  // A-fragments of Q (reused by rel and by scores; K = DQ = 32 -> one step).
  v16bf aq[2];
#pragma unroll
  for (int i = 0; i < 2; ++i) {
    const bf16* row = sQ + (i * 16 + l15) * 40;
    aq[i] = frag_cat(row + hi * 16, row + hi * 16 + 8);
  }

  // rel_dot[32][208] = Q x rel^T  (13 column tiles over 8 waves)
  for (int ct = wid; ct < 13; ct += 8) {
    const bf16* bp = prel_l + ((size_t)h * NRELP + ct * 16 + l15) * DQ + hi * 16;
    v16bf bfr = frag_cat(bp, bp + 8);
#pragma unroll
    for (int i = 0; i < 2; ++i) {
      v8f c = {};
      c = wmma_bf16(aq[i], bfr, c);
#pragma unroll
      for (int r = 0; r < 8; ++r)
        sRel[(i * 16 + r + hi * 8) * NRELP + ct * 16 + l15] = c[r];
    }
  }
  __syncthreads();

  // Scores: wave w owns key columns [w*64, w*64+64) of all 32 query rows.
  v8f acc[2][4] = {};
  const int nW = wid * 64;
#pragma unroll
  for (int j = 0; j < 4; ++j) {
    const bf16* bp = sKP + (nW + j * 16 + l15) * 40 + hi * 16;
    v16bf bk = frag_cat(bp, bp + 8);
#pragma unroll
    for (int i = 0; i < 2; ++i)
      acc[i][j] = wmma_bf16(aq[i], bk, acc[i][j]);
  }

  // Add relative-position logits (d = k - q + 99, mask |k-q| > 99) + row max.
#pragma unroll
  for (int i = 0; i < 2; ++i)
#pragma unroll
    for (int r = 0; r < 8; ++r) {
      const int rowL = i * 16 + r + hi * 8;
      const int rowA = qBase + rowL;
      float m = -3.0e38f;
#pragma unroll
      for (int j = 0; j < 4; ++j) {
        int col = nW + j * 16 + l15;
        int d = col - rowA + 99;
        float add = (d >= 0 && d < NREL) ? sRel[rowL * NRELP + d] : -1.0e8f;
        float v = acc[i][j][r] + add;
        acc[i][j][r] = v;
        m = fmaxf(m, v);
      }
      m = fmaxf(m, __shfl_xor(m, 1, 32));
      m = fmaxf(m, __shfl_xor(m, 2, 32));
      m = fmaxf(m, __shfl_xor(m, 4, 32));
      m = fmaxf(m, __shfl_xor(m, 8, 32));
      if (l15 == 0) sRedM[wid * 32 + rowL] = m;
    }
  __syncthreads();

  // Global row max -> exp -> partial row sums.
#pragma unroll
  for (int i = 0; i < 2; ++i)
#pragma unroll
    for (int r = 0; r < 8; ++r) {
      const int rowL = i * 16 + r + hi * 8;
      float gm = sRedM[rowL];
#pragma unroll
      for (int w2 = 1; w2 < 8; ++w2) gm = fmaxf(gm, sRedM[w2 * 32 + rowL]);
      float s = 0.0f;
#pragma unroll
      for (int j = 0; j < 4; ++j) {
        float e = __expf(acc[i][j][r] - gm);
        acc[i][j][r] = e;
        s += e;
      }
      s += __shfl_xor(s, 1, 32);
      s += __shfl_xor(s, 2, 32);
      s += __shfl_xor(s, 4, 32);
      s += __shfl_xor(s, 8, 32);
      if (l15 == 0) sRedS[wid * 32 + rowL] = s;
    }
  for (int e = tid; e < 32 * 32; e += 256) sO[e] = 0.0f;
  __syncthreads();

  // Normalize, write probs as bf16 into the (now dead) K-tile region.
#pragma unroll
  for (int i = 0; i < 2; ++i)
#pragma unroll
    for (int r = 0; r < 8; ++r) {
      const int rowL = i * 16 + r + hi * 8;
      float gs = sRedS[rowL];
#pragma unroll
      for (int w2 = 1; w2 < 8; ++w2) gs += sRedS[w2 * 32 + rowL];
      float inv = 1.0f / gs;
#pragma unroll
      for (int j = 0; j < 4; ++j)
        sKP[rowL * 520 + nW + j * 16 + l15] = (bf16)(acc[i][j][r] * inv);
    }
  __syncthreads();

  // O = probs(32x512) x V(512x32); K split across waves (64 each).
  v8f oacc[2][2] = {};
#pragma unroll
  for (int ks = 0; ks < 2; ++ks) {
    const int k0 = wid * 64 + ks * 32;
    v16bf ap[2];
#pragma unroll
    for (int i = 0; i < 2; ++i) {
      const bf16* row = sKP + (i * 16 + l15) * 520 + k0;
      ap[i] = frag_cat(row + hi * 16, row + hi * 16 + 8);
    }
#pragma unroll
    for (int j = 0; j < 2; ++j) {
      const bf16* bp = sVT + (j * 16 + l15) * 520 + k0 + hi * 16;
      v16bf bv = frag_cat(bp, bp + 8);
#pragma unroll
      for (int i = 0; i < 2; ++i)
        oacc[i][j] = wmma_bf16(ap[i], bv, oacc[i][j]);
    }
  }
#pragma unroll
  for (int i = 0; i < 2; ++i)
#pragma unroll
    for (int j = 0; j < 2; ++j)
#pragma unroll
      for (int r = 0; r < 8; ++r)
        atomicAdd(&sO[(i * 16 + r + hi * 8) * 32 + j * 16 + l15],
                  oacc[i][j][r]);  // ds_add_f32 cross-wave reduction
  __syncthreads();

  for (int e = tid; e < 32 * 32; e += 256) {
    int r = e >> 5, a = e & 31;
    ob[(size_t)(b * TT + qBase + r) * 256 + h * DQ + a] = (bf16)sO[e];
  }
}

// ---------------------------------------------------------------------------
// Residual add + LayerNorm (+ bf16 re-cast of activations). One block per row.
// ---------------------------------------------------------------------------
__global__ __launch_bounds__(256)
void ln_kernel(float* __restrict__ h, const float* __restrict__ add,
               const float* __restrict__ w, const float* __restrict__ b,
               bf16* __restrict__ xb) {
  __shared__ float red[8];
  const int row = blockIdx.x;
  const int tid = threadIdx.x;
  float v[3];
#pragma unroll
  for (int i = 0; i < 3; ++i) {
    size_t idx = (size_t)row * DM + tid + i * 256;
    v[i] = h[idx] + add[idx];
  }
  float s = v[0] + v[1] + v[2];
#pragma unroll
  for (int m = 1; m < 32; m <<= 1) s += __shfl_xor(s, m, 32);
  if ((tid & 31) == 0) red[tid >> 5] = s;
  __syncthreads();
  float tot = 0.0f;
#pragma unroll
  for (int i = 0; i < 8; ++i) tot += red[i];
  const float mu = tot * (1.0f / DM);
  __syncthreads();
  float sv = 0.0f;
#pragma unroll
  for (int i = 0; i < 3; ++i) { float d = v[i] - mu; sv += d * d; }
#pragma unroll
  for (int m = 1; m < 32; m <<= 1) sv += __shfl_xor(sv, m, 32);
  if ((tid & 31) == 0) red[tid >> 5] = sv;
  __syncthreads();
  float var = 0.0f;
#pragma unroll
  for (int i = 0; i < 8; ++i) var += red[i];
  var *= (1.0f / DM);
  const float rs = rsqrtf(var + 1e-5f);
#pragma unroll
  for (int i = 0; i < 3; ++i) {
    int c = tid + i * 256;
    size_t idx = (size_t)row * DM + c;
    float y = (v[i] - mu) * rs * w[c] + b[c];
    h[idx] = y;
    xb[idx] = (bf16)y;
  }
}

// ---------------------------------------------------------------------------
// Host orchestration
// ---------------------------------------------------------------------------
extern "C" void kernel_launch(void* const* d_in, const int* in_sizes, int n_in,
                              void* d_out, int out_size, void* d_ws,
                              size_t ws_size, hipStream_t stream) {
  (void)in_sizes; (void)n_in; (void)out_size; (void)ws_size;
  const float* x      = (const float*)d_in[0];
  const float* wq     = (const float*)d_in[1];
  const float* wk     = (const float*)d_in[2];
  const float* wv     = (const float*)d_in[3];
  const float* wo     = (const float*)d_in[4];
  const float* rel    = (const float*)d_in[5];
  const float* lin1_w = (const float*)d_in[6];
  const float* lin1_b = (const float*)d_in[7];
  const float* lin2_w = (const float*)d_in[8];
  const float* lin2_b = (const float*)d_in[9];
  const float* ln1_w  = (const float*)d_in[10];
  const float* ln1_b  = (const float*)d_in[11];
  const float* ln2_w  = (const float*)d_in[12];
  const float* ln2_b  = (const float*)d_in[13];
  const float* w_out  = (const float*)d_in[14];
  const float* b_out  = (const float*)d_in[15];

  char* ws = (char*)d_ws;
  size_t off = 0;
  auto take = [&](size_t bytes) -> char* {
    char* p = ws + off;
    off = (off + bytes + 255) & ~(size_t)255;
    return p;
  };
  bf16*  pqkv  = (bf16*)take((size_t)LNUM * 768 * DM * 2);
  bf16*  pwo   = (bf16*)take((size_t)LNUM * DM * 256 * 2);
  bf16*  prel  = (bf16*)take((size_t)LNUM * HH * NRELP * DQ * 2);
  bf16*  pl1   = (bf16*)take((size_t)LNUM * FF * DM * 2);
  bf16*  pl2   = (bf16*)take((size_t)LNUM * DM * FF * 2);
  bf16*  pwout = (bf16*)take((size_t)128 * DM * 2);
  float* hbuf  = (float*)take((size_t)BT * DM * 4);
  bf16*  xbf   = (bf16*)take((size_t)BT * DM * 2);
  bf16*  qkvb  = (bf16*)take((size_t)BT * 768 * 2);
  bf16*  obuf  = (bf16*)take((size_t)BT * 256 * 2);
  float* tmp   = (float*)take((size_t)BT * DM * 4);
  bf16*  ffm   = (bf16*)take((size_t)BT * FF * 2);

  auto blocks = [](size_t n) {
    size_t bl = (n + 255) / 256;
    return (unsigned)(bl > 32768 ? 32768 : bl);
  };

  // One-time (per launch) weight packing: f32 -> bf16 [N][K].
  pack_qkv_kernel<<<blocks((size_t)LNUM * 768 * DM), 256, 0, stream>>>(
      wq, wk, wv, pqkv);
  pack_wo_kernel<<<blocks((size_t)LNUM * DM * 256), 256, 0, stream>>>(wo, pwo);
  pack_rel_kernel<<<blocks((size_t)LNUM * HH * NRELP * DQ), 256, 0, stream>>>(
      rel, prel);
  conv_kernel<<<blocks((size_t)LNUM * FF * DM), 256, 0, stream>>>(
      lin1_w, pl1, (size_t)LNUM * FF * DM);
  conv_kernel<<<blocks((size_t)LNUM * DM * FF), 256, 0, stream>>>(
      lin2_w, pl2, (size_t)LNUM * DM * FF);
  pack_wout_kernel<<<blocks((size_t)128 * DM), 256, 0, stream>>>(w_out, pwout);
  init_h_kernel<<<blocks((size_t)BT * DM), 256, 0, stream>>>(
      x, hbuf, xbf, (size_t)BT * DM);

  const float qscale = 0.17677669529663687f;  // 1/sqrt(DQ)
  for (int l = 0; l < LNUM; ++l) {
    // Fused QKV projection (M=2048, N=768, K=768) -> bf16 [BT][768].
    gemm_bf16_kernel<<<dim3(6, BT / 128), 256, 0, stream>>>(
        xbf, DM, pqkv + (size_t)l * 768 * DM, DM, nullptr, qkvb, 768, 768, DM,
        nullptr, 0);
    // Attention with relative-position logits + softmax + probs@V.
    attn_kernel<<<dim3(TT / 32, BB * HH), 256, SMEM_ATTN, stream>>>(
        qkvb, prel + (size_t)l * HH * NRELP * DQ, obuf, qscale);
    // Output projection (M=2048, N=768, K=256) -> tmp f32.
    gemm_bf16_kernel<<<dim3(6, BT / 128), 256, 0, stream>>>(
        obuf, 256, pwo + (size_t)l * DM * 256, 256, tmp, nullptr, DM, DM, 256,
        nullptr, 0);
    ln_kernel<<<BT, 256, 0, stream>>>(hbuf, tmp, ln1_w + l * DM,
                                      ln1_b + l * DM, xbf);
    // FFN: 768 -> 3072 (ReLU, bf16 out) -> 768 (f32 out).
    gemm_bf16_kernel<<<dim3(24, BT / 128), 256, 0, stream>>>(
        xbf, DM, pl1 + (size_t)l * FF * DM, DM, nullptr, ffm, FF, FF, DM,
        lin1_b + (size_t)l * FF, 1);
    gemm_bf16_kernel<<<dim3(6, BT / 128), 256, 0, stream>>>(
        ffm, FF, pl2 + (size_t)l * DM * FF, FF, tmp, nullptr, DM, DM, FF,
        lin2_b + (size_t)l * DM, 0);
    ln_kernel<<<BT, 256, 0, stream>>>(hbuf, tmp, ln2_w + l * DM,
                                      ln2_b + l * DM, xbf);
  }
  // Final projection to 80 outputs (N padded to 128 in packed weights).
  gemm_bf16_kernel<<<dim3(1, BT / 128), 256, 0, stream>>>(
      xbf, DM, pwout, DM, (float*)d_out, nullptr, OUTN, OUTN, DM, b_out, 0);
}
